// THDStack_30219389894928
// MI455X (gfx1250) — matrix operand order repaired and checked by reading, not compiled, not run on hardware
//
#include <hip/hip_runtime.h>
#include <math.h>

typedef __attribute__((ext_vector_type(16))) _Float16 v16h;
typedef __attribute__((ext_vector_type(8)))  _Float16 v8h;
typedef __attribute__((ext_vector_type(8)))  float    v8f;
typedef __attribute__((ext_vector_type(4)))  float    v4f;
typedef __attribute__((ext_vector_type(4)))  unsigned int v4u;
typedef __attribute__((ext_vector_type(8)))  unsigned int v8u;

#define LN_EPS 1e-5f

// =====================================================================
// Kernel 1: conv3x3(SAME) + bias + LayerNorm(D=128) + ReLU
// One workgroup = 16 consecutive pixels of one image row. 128 threads,
// thread index == output channel. Input patch (64ch x 3 x 18) in LDS.
// Writes token-major t [B, N=4096, D=128] in both f32 and f16.
// =====================================================================
__global__ __launch_bounds__(128)
void conv_ln_relu_kernel(const float* __restrict__ x,    // [B,64,64,64]
                         const float* __restrict__ cw,   // [128,64,3,3]
                         const float* __restrict__ cb,   // [128]
                         const float* __restrict__ lw,   // [128]
                         const float* __restrict__ lb,   // [128]
                         float*    __restrict__ tf,      // [B,4096,128]
                         _Float16* __restrict__ th)      // [B,4096,128]
{
    const int H = 64, W = 64, C = 64;
    int wg = blockIdx.x;                 // b*256 + h*4 + (w0/16)
    int b  = wg >> 8;
    int r  = wg & 255;
    int h  = r >> 2;
    int w0 = (r & 3) * 16;
    int tid = threadIdx.x;               // output channel

    __shared__ float patch[64][3][18];   // [c][ky][x], x covers w0-1 .. w0+16
    __shared__ float vals[16][128];      // [pixel][channel]

    // cooperative patch load with zero padding
    for (int e = tid; e < C * 3 * 18; e += 128) {
        int c  = e / 54;
        int rr = e % 54;
        int ky = rr / 18;
        int xx = rr % 18;
        int gh = h + ky - 1;
        int gw = w0 + xx - 1;
        float v = 0.0f;
        if (gh >= 0 && gh < H && gw >= 0 && gw < W)
            v = x[(((size_t)b * C + c) * H + gh) * W + gw];
        patch[c][ky][xx] = v;
    }
    __syncthreads();

    float acc[16];
    float bias = cb[tid];
#pragma unroll
    for (int p = 0; p < 16; ++p) acc[p] = bias;

    for (int c = 0; c < C; ++c) {
#pragma unroll
        for (int ky = 0; ky < 3; ++ky) {
            float pr[18];
#pragma unroll
            for (int xx = 0; xx < 18; ++xx) pr[xx] = patch[c][ky][xx];
#pragma unroll
            for (int kx = 0; kx < 3; ++kx) {
                float wv = cw[(((size_t)tid * C + c) * 3 + ky) * 3 + kx];
#pragma unroll
                for (int p = 0; p < 16; ++p) acc[p] += pr[p + kx] * wv;
            }
        }
    }

#pragma unroll
    for (int p = 0; p < 16; ++p) vals[p][tid] = acc[p];
    __syncthreads();

    // LayerNorm + ReLU: one wave per token, 4 tokens per wave, 4 ch per lane
    int wave = tid >> 5, lane = tid & 31;
    for (int i = 0; i < 4; ++i) {
        int t = wave * 4 + i;
        float v0 = vals[t][lane * 4 + 0];
        float v1 = vals[t][lane * 4 + 1];
        float v2 = vals[t][lane * 4 + 2];
        float v3 = vals[t][lane * 4 + 3];
        float s = v0 + v1 + v2 + v3;
#pragma unroll
        for (int off = 16; off >= 1; off >>= 1) s += __shfl_xor(s, off);
        float mu = s * (1.0f / 128.0f);
        float d0 = v0 - mu, d1 = v1 - mu, d2 = v2 - mu, d3 = v3 - mu;
        float q = d0 * d0 + d1 * d1 + d2 * d2 + d3 * d3;
#pragma unroll
        for (int off = 16; off >= 1; off >>= 1) q += __shfl_xor(q, off);
        float rstd = rsqrtf(q * (1.0f / 128.0f) + LN_EPS);

        size_t tok = ((size_t)b * H + h) * W + (w0 + t);
        float dv[4] = { d0, d1, d2, d3 };
#pragma unroll
        for (int j = 0; j < 4; ++j) {
            int c = lane * 4 + j;
            float y = dv[j] * rstd * lw[c] + lb[c];
            y = fmaxf(y, 0.0f);
            tf[tok * 128 + c] = y;
            th[tok * 128 + c] = (_Float16)y;
        }
    }
}

// =====================================================================
// Kernel 2: flash attention  O = softmax(Q K^T * scale) V  (per batch)
// 128 threads = 4 waves; each wave owns 16 query rows (64 per WG).
// Key/value blocks of 64: K staged row-major via the Tensor Data Mover
// (tensor_load_to_lds), V staged transposed by VALU. All matmuls via
// v_wmma_f32_16x16x32_f16; online softmax state kept in C/D layout.
// =====================================================================
__global__ __launch_bounds__(128)
void flash_attn_kernel(const _Float16* __restrict__ Q,
                       const _Float16* __restrict__ K,
                       const _Float16* __restrict__ V,
                       float* __restrict__ O,
                       int Ntok, float scale)
{
    const int Dm = 128;
    int qblk = blockIdx.x;               // 64 queries per block
    int b    = blockIdx.y;
    int tid  = threadIdx.x;
    int wave = tid >> 5, lane = tid & 31;
    int half = lane >> 4, l16 = lane & 15;
    int kbase = half * 8;
    size_t base = (size_t)b * Ntok * Dm;

    __shared__ __attribute__((aligned(32))) _Float16 Kb[64][128];   // [key][dim]
    __shared__ __attribute__((aligned(32))) _Float16 Vt[128][64];   // [dim][key]
    __shared__ __attribute__((aligned(32))) _Float16 Pst[4][16][64];

    // ---- Q A-fragments: 16 rows x 128 dims = 4 chunks of 16x32 ----
    int qrow = qblk * 64 + wave * 16 + l16;
    v16h qa[4];
#pragma unroll
    for (int kc = 0; kc < 4; ++kc) {
        const _Float16* qp = Q + base + (size_t)qrow * Dm + kc * 32 + kbase;
        v8h lo = *(const v8h*)qp;
        v8h hi = *(const v8h*)(qp + 16);
#pragma unroll
        for (int i = 0; i < 8; ++i) { qa[kc][i] = lo[i]; qa[kc][8 + i] = hi[i]; }
    }

    v8f accum[8];
#pragma unroll
    for (int d = 0; d < 8; ++d)
#pragma unroll
        for (int i = 0; i < 8; ++i) accum[d][i] = 0.0f;
    float mrow[8], lrow[8];
#pragma unroll
    for (int v = 0; v < 8; ++v) { mrow[v] = -INFINITY; lrow[v] = 0.0f; }

    for (int kb0 = 0; kb0 < Ntok; kb0 += 64) {
        __syncthreads();   // previous block fully consumed before restaging

        // ---- K block via Tensor Data Mover: 64x128 f16 tile -> LDS ----
        if (wave == 0) {
            unsigned long long ga =
                (unsigned long long)(uintptr_t)(K + base + (size_t)kb0 * Dm);
            v4u g0;
            g0[0] = 1u;                                  // count=1, user mode
            g0[1] = (unsigned int)(uintptr_t)(&Kb[0][0]);// lds_addr
            g0[2] = (unsigned int)ga;                    // global_addr[31:0]
            g0[3] = (unsigned int)(ga >> 32) | 0x80000000u; // ga[56:32] | type=2
            v8u g1;
            g1[0] = 1u << 16;                            // data_size=2B
            g1[1] = 128u << 16;                          // tensor_dim0 = 128
            g1[2] = ((unsigned int)Ntok & 0xffffu) << 16;// tensor_dim1 lo
            g1[3] = (((unsigned int)Ntok >> 16) & 0xffffu) | (128u << 16); // dim1 hi | tile_dim0=128
            g1[4] = 64u;                                 // tile_dim1 = 64 rows
            g1[5] = 128u;                                // tensor_dim0_stride
            g1[6] = 0u;
            g1[7] = 0u;
            asm volatile("tensor_load_to_lds %0, %1" :: "s"(g0), "s"(g1) : "memory");
        }

        // ---- V block staged transposed: Vt[dim][key] ----
        {
            int r  = tid >> 1;             // key row within block
            int qd = (tid & 1) * 64;       // dim range start
            const _Float16* vp = V + base + (size_t)(kb0 + r) * Dm + qd;
#pragma unroll
            for (int j = 0; j < 64; j += 8) {
                v8h vv = *(const v8h*)(vp + j);
#pragma unroll
                for (int i = 0; i < 8; ++i) Vt[qd + j + i][r] = vv[i];
            }
            if (kb0 + 64 < Ntok)   // pull next V block toward caches
                __builtin_prefetch(V + base + (size_t)(kb0 + 64 + r) * Dm + qd, 0, 1);
        }

        if (wave == 0) __builtin_amdgcn_s_wait_tensorcnt(0);
        __syncthreads();

        // ---- S = (Q K^T) * scale : four 16x16 tiles per wave ----
        v8f s[4];
#pragma unroll
        for (int t = 0; t < 4; ++t) {
            v16h kf[4];
#pragma unroll
            for (int kc = 0; kc < 4; ++kc) {
                const _Float16* kp = &Kb[t * 16 + l16][kc * 32 + kbase];
                v8h lo = *(const v8h*)kp;
                v8h hi = *(const v8h*)(kp + 16);
#pragma unroll
                for (int i = 0; i < 8; ++i) { kf[kc][i] = lo[i]; kf[kc][8 + i] = hi[i]; }
            }
            v8f c;
#pragma unroll
            for (int i = 0; i < 8; ++i) c[i] = 0.0f;
#pragma unroll
            for (int kc = 0; kc < 4; ++kc)
                c = __builtin_amdgcn_wmma_f32_16x16x32_f16(
                        false, qa[kc], false, kf[kc], (short)0, c, false, false);
            s[t] = c;
        }

        // ---- online softmax (row = v + 8*half; 16 lanes = 16 cols) ----
        float alpha[8];
#pragma unroll
        for (int v = 0; v < 8; ++v) {
            float sv[4];
#pragma unroll
            for (int t = 0; t < 4; ++t) sv[t] = s[t][v] * scale;
            float mx = fmaxf(fmaxf(sv[0], sv[1]), fmaxf(sv[2], sv[3]));
#pragma unroll
            for (int off = 8; off >= 1; off >>= 1) mx = fmaxf(mx, __shfl_xor(mx, off));
            float mnew = fmaxf(mrow[v], mx);
            float a = __expf(mrow[v] - mnew);
            float ps = 0.0f;
#pragma unroll
            for (int t = 0; t < 4; ++t) { sv[t] = __expf(sv[t] - mnew); ps += sv[t]; }
#pragma unroll
            for (int off = 8; off >= 1; off >>= 1) ps += __shfl_xor(ps, off);
            lrow[v] = lrow[v] * a + ps;
            mrow[v] = mnew;
            alpha[v] = a;
#pragma unroll
            for (int t = 0; t < 4; ++t) s[t][v] = sv[t];
        }
#pragma unroll
        for (int d = 0; d < 8; ++d)
#pragma unroll
            for (int v = 0; v < 8; ++v) accum[d][v] *= alpha[v];

        // ---- restage P (C/D layout -> A layout) via per-wave LDS ----
#pragma unroll
        for (int t = 0; t < 4; ++t)
#pragma unroll
            for (int v = 0; v < 8; ++v)
                Pst[wave][v + 8 * half][t * 16 + l16] = (_Float16)s[t][v];
        asm volatile("s_wait_dscnt 0" ::: "memory");   // same-wave store->load

        v16h pa[2];
#pragma unroll
        for (int pc = 0; pc < 2; ++pc) {
            const _Float16* pp = &Pst[wave][l16][pc * 32 + kbase];
            v8h lo = *(const v8h*)pp;
            v8h hi = *(const v8h*)(pp + 16);
#pragma unroll
            for (int i = 0; i < 8; ++i) { pa[pc][i] = lo[i]; pa[pc][8 + i] = hi[i]; }
        }

        // ---- accum += P @ V : 8 dim-tiles x 2 K-chunks of 32 ----
#pragma unroll
        for (int d = 0; d < 8; ++d) {
            v16h vf[2];
#pragma unroll
            for (int pc = 0; pc < 2; ++pc) {
                const _Float16* vp2 = &Vt[d * 16 + l16][pc * 32 + kbase];
                v8h lo = *(const v8h*)vp2;
                v8h hi = *(const v8h*)(vp2 + 16);
#pragma unroll
                for (int i = 0; i < 8; ++i) { vf[pc][i] = lo[i]; vf[pc][8 + i] = hi[i]; }
            }
            accum[d] = __builtin_amdgcn_wmma_f32_16x16x32_f16(
                           false, pa[0], false, vf[0], (short)0, accum[d], false, false);
            accum[d] = __builtin_amdgcn_wmma_f32_16x16x32_f16(
                           false, pa[1], false, vf[1], (short)0, accum[d], false, false);
        }
    }

    // ---- finalize: O = accum / l ----
#pragma unroll
    for (int d = 0; d < 8; ++d) {
#pragma unroll
        for (int v = 0; v < 8; ++v) {
            int row = qblk * 64 + wave * 16 + v + 8 * half;
            O[base + (size_t)row * Dm + d * 16 + l16] = accum[d][v] / lrow[v];
        }
    }
}

// =====================================================================
// Kernel 3: out = LayerNorm(t + x_attn). One wave per token (4/WG).
// =====================================================================
__global__ __launch_bounds__(128)
void residual_ln_kernel(const float* __restrict__ t,
                        const float* __restrict__ xa,
                        const float* __restrict__ lw,
                        const float* __restrict__ lb,
                        float* __restrict__ out)
{
    int wave = threadIdx.x >> 5, lane = threadIdx.x & 31;
    size_t tok = (size_t)blockIdx.x * 4 + wave;
    size_t off = tok * 128 + lane * 4;
    v4f tv = *(const v4f*)(t + off);
    v4f xv = *(const v4f*)(xa + off);
    float v[4];
#pragma unroll
    for (int j = 0; j < 4; ++j) v[j] = tv[j] + xv[j];
    float s = v[0] + v[1] + v[2] + v[3];
#pragma unroll
    for (int o = 16; o >= 1; o >>= 1) s += __shfl_xor(s, o);
    float mu = s * (1.0f / 128.0f);
    float q = 0.0f;
#pragma unroll
    for (int j = 0; j < 4; ++j) { v[j] -= mu; q += v[j] * v[j]; }
#pragma unroll
    for (int o = 16; o >= 1; o >>= 1) q += __shfl_xor(q, o);
    float rstd = rsqrtf(q * (1.0f / 128.0f) + LN_EPS);
#pragma unroll
    for (int j = 0; j < 4; ++j) {
        int c = lane * 4 + j;
        out[off + j] = v[j] * rstd * lw[c] + lb[c];
    }
}

// =====================================================================
// Host launcher
// =====================================================================
extern "C" void kernel_launch(void* const* d_in, const int* in_sizes, int n_in,
                              void* d_out, int out_size, void* d_ws, size_t ws_size,
                              hipStream_t stream) {
    (void)in_sizes; (void)n_in; (void)out_size; (void)ws_size;
    const float* x1   = (const float*)d_in[0];
    const float* x2   = (const float*)d_in[1];
    const float* c1w  = (const float*)d_in[2];
    const float* c1b  = (const float*)d_in[3];
    const float* c2w  = (const float*)d_in[4];
    const float* c2b  = (const float*)d_in[5];
    const float* ln1w = (const float*)d_in[6];
    const float* ln1b = (const float*)d_in[7];
    const float* ln2w = (const float*)d_in[8];
    const float* ln2b = (const float*)d_in[9];
    const float* ln3w = (const float*)d_in[10];
    const float* ln3b = (const float*)d_in[11];
    const float* ln4w = (const float*)d_in[12];
    const float* ln4b = (const float*)d_in[13];

    const int B = 8, N = 4096, D = 128;
    const size_t BND = (size_t)B * N * D;

    // workspace: 4 f32 tensors + 2 f16 tensors = ~84 MB
    float*    wsf = (float*)d_ws;
    float*    t1f = wsf;
    float*    t2f = wsf + BND;
    float*    x1a = wsf + 2 * BND;
    float*    x2a = wsf + 3 * BND;
    _Float16* t1h = (_Float16*)(wsf + 4 * BND);
    _Float16* t2h = t1h + BND;
    float*    out = (float*)d_out;

    // 1) conv + LN + ReLU -> tokens (f32 + f16)
    conv_ln_relu_kernel<<<dim3(2048), dim3(128), 0, stream>>>(x1, c1w, c1b, ln1w, ln1b, t1f, t1h);
    conv_ln_relu_kernel<<<dim3(2048), dim3(128), 0, stream>>>(x2, c2w, c2b, ln2w, ln2b, t2f, t2h);

    // 2) flash attention (WMMA): x1_ = softmax(t2 t2^T)*t1 ; x2_ = softmax(t1 t1^T)*t2
    const float scale = 0.08838834764831845f;  // 1/sqrt(128)
    dim3 fgrid(N / 64, B);
    flash_attn_kernel<<<fgrid, dim3(128), 0, stream>>>(t2h, t2h, t1h, x1a, N, scale);
    flash_attn_kernel<<<fgrid, dim3(128), 0, stream>>>(t1h, t1h, t2h, x2a, N, scale);

    // 3) residual + LayerNorm -> d_out (o1 || o2)
    residual_ln_kernel<<<dim3(B * N / 4), dim3(128), 0, stream>>>(t1f, x1a, ln3w, ln3b, out);
    residual_ln_kernel<<<dim3(B * N / 4), dim3(128), 0, stream>>>(t2f, x2a, ln4w, ln4b, out + BND);
}
